// MultiHeadAttention_13855564497653
// MI455X (gfx1250) — matrix operand order repaired and checked by reading, not compiled, not run on hardware
//
#include <hip/hip_runtime.h>
#include <hip/hip_bf16.h>
#include <math.h>

// ---------------------------------------------------------------------------
// MI455X (gfx1250) multi-head attention, bf16 WMMA everywhere.
//   B=4, T=1024, E=D=768, H=8.  ~258 GFLOP vs ~100MB traffic -> compute bound:
//   every matmul stage runs through v_wmma_f32_16x16x32_bf16 (wave32), bf16
//   conversion fused into tile staging. Async global->LDS (ASYNCcnt) used for
//   the pure-copy bf16 tile fills; packed u32 DS stores for converted tiles.
// Workspace layout (requires ws_size >= ~202 MB):
//   q,k,v : bf16 [B,H,T,D]      (3 x 50,331,648 B)
//   z     : bf16 [B,T,H*D]      (50,331,648 B)
//   bsum  : f32  [E]            (3,072 B)
// ---------------------------------------------------------------------------

#define B_ 4
#define T_ 1024
#define E_ 768
#define D_ 768
#define H_ 8

typedef __attribute__((ext_vector_type(16))) __bf16 v16bf;
typedef __attribute__((ext_vector_type(8)))  float  v8f;

static_assert(sizeof(v16bf) == 32, "v16bf must be 32 bytes");

union FragBF { v16bf v; unsigned int u[8]; };

__device__ __forceinline__ unsigned short f32_to_bf16(float f) {
  unsigned int u = __float_as_uint(f);
  u += 0x7fffu + ((u >> 16) & 1u);   // round-to-nearest-even
  return (unsigned short)(u >> 16);
}

__device__ __forceinline__ unsigned int pack_bf16x2(float lo, float hi) {
  return (unsigned int)f32_to_bf16(lo) | ((unsigned int)f32_to_bf16(hi) << 16);
}

// Low 32 bits of a generic pointer to LDS == LDS byte address (aperture rule).
__device__ __forceinline__ unsigned int lds_addr32(const void* p) {
  return (unsigned int)(unsigned long long)p;
}

// CDNA5 async global->LDS copy, 16B per lane, tracked by ASYNCcnt.
__device__ __forceinline__ void async_copy_b128(const void* gptr, unsigned int lds_off) {
  asm volatile("global_load_async_to_lds_b128 %0, %1, off"
               :: "v"(lds_off), "v"(gptr) : "memory");
}
__device__ __forceinline__ void wait_async0() {
  asm volatile("s_wait_asynccnt 0" ::: "memory");
}

// A-matrix fragment (16x32 bf16, M x K), LDS tile laid [m][k], k contiguous.
// ISA layout: lanes 0-15 hold M=lane, K={0..7,16..23}; lanes 16-31 K={8..15,24..31}.
__device__ __forceinline__ v16bf load_A(const unsigned short* lds, int m_base, int ld) {
  const int lane = threadIdx.x & 31;
  const int koff = (lane < 16) ? 0 : 8;
  const unsigned short* row = lds + (m_base + (lane & 15)) * ld;
  FragBF f;
#pragma unroll
  for (int r = 0; r < 8; ++r) {
    int j = 2 * r;
    int k = ((j < 8) ? j : j + 8) + koff;          // contiguous pair (k,k+1)
    f.u[r] = *(const unsigned int*)(row + k);
  }
  return f.v;
}

// B-matrix fragment (32x16 bf16, K x N), LDS tile laid [n][k], k contiguous.
// Lanes 0-15: N=lane, K={0..7,16..23}; lanes 16-31: N=lane-16, K={8..15,24..31}.
__device__ __forceinline__ v16bf load_B(const unsigned short* lds, int n_base, int ld) {
  const int lane = threadIdx.x & 31;
  const int kbase = (lane < 16) ? 0 : 8;
  const unsigned short* row = lds + (n_base + (lane & 15)) * ld + kbase;
  FragBF f;
#pragma unroll
  for (int r = 0; r < 8; ++r) f.u[r] = *(const unsigned int*)(row + ((r < 4) ? 2 * r : 2 * r + 8));
  return f.v;
}

__device__ __forceinline__ v8f wmma_bf16(v16bf a, v16bf b, v8f c) {
  return __builtin_amdgcn_wmma_f32_16x16x32_bf16(
      /*neg_a=*/false, a, /*neg_b=*/false, b,
      /*c_mod=*/(short)0, c, /*reuse_a=*/false, /*reuse_b=*/false);
}

// ---------------------------------------------------------------------------
// Q/K/V projection: Y = X(4096x768) * W_h(768x768) + b_h -> bf16 [B,H,T,D]
// grid (M/128, D/128, H), block 256 (8 waves: 4 M x 2 N), tile 128x128, K=32
// Per K-step per wave: 6 fragment loads, 8 WMMAs.
// ---------------------------------------------------------------------------
__global__ __launch_bounds__(256)
void proj_gemm(const float* __restrict__ X, const float* __restrict__ W,
               const float* __restrict__ bias, unsigned short* __restrict__ Y) {
  __shared__ alignas(16) unsigned short As[128 * 32];   // [m][k]
  __shared__ alignas(16) unsigned short Bs[128 * 32];   // [n][k] (transposed)
  const int h   = blockIdx.z;
  const int m0  = blockIdx.x * 128;
  const int n0  = blockIdx.y * 128;
  const int tid = threadIdx.x;
  const int w = tid >> 5, lane = tid & 31;
  const int wm = w >> 1;   // 0..3 -> 32 rows each
  const int wn = w & 1;    // 0..1 -> 64 cols each
  const float* Wh = W + (size_t)h * E_ * D_;

  const v8f vzero = {0.f, 0.f, 0.f, 0.f, 0.f, 0.f, 0.f, 0.f};
  v8f acc[2][4];
#pragma unroll
  for (int mi = 0; mi < 2; ++mi)
#pragma unroll
    for (int ni = 0; ni < 4; ++ni) acc[mi][ni] = vzero;

  for (int k0 = 0; k0 < E_; k0 += 32) {
    if (k0 + 32 < E_) {  // speculative prefetch of next K tile (global_prefetch_b8)
      __builtin_prefetch(&X[(size_t)(m0 + (tid >> 1)) * E_ + k0 + 32], 0, 1);
      __builtin_prefetch(&Wh[(size_t)(k0 + 32 + (tid & 31)) * D_ + n0], 0, 1);
    }
    // A: 128x32 f32 -> bf16, packed u32 stores (2048 pairs / 256 threads)
#pragma unroll
    for (int e = 0; e < 8; ++e) {
      int p = e * 256 + tid;
      int i = p >> 4, k2 = (p & 15) * 2;
      float2 xv = *(const float2*)&X[(size_t)(m0 + i) * E_ + k0 + k2];
      *(unsigned int*)&As[i * 32 + k2] = pack_bf16x2(xv.x, xv.y);
    }
    // B: 32x128 -> transposed [n][k], packed along k (2048 pairs)
#pragma unroll
    for (int e = 0; e < 8; ++e) {
      int p = e * 256 + tid;
      int n = p & 127, kk = (p >> 7) * 2;
      float blo = Wh[(size_t)(k0 + kk) * D_ + n0 + n];
      float bhi = Wh[(size_t)(k0 + kk + 1) * D_ + n0 + n];
      *(unsigned int*)&Bs[n * 32 + kk] = pack_bf16x2(blo, bhi);
    }
    __syncthreads();
    v16bf afr[2], bfr[4];
#pragma unroll
    for (int mi = 0; mi < 2; ++mi) afr[mi] = load_A(As, wm * 32 + mi * 16, 32);
#pragma unroll
    for (int ni = 0; ni < 4; ++ni) bfr[ni] = load_B(Bs, wn * 64 + ni * 16, 32);
#pragma unroll
    for (int mi = 0; mi < 2; ++mi)
#pragma unroll
      for (int ni = 0; ni < 4; ++ni)
        acc[mi][ni] = wmma_bf16(afr[mi], bfr[ni], acc[mi][ni]);
    __syncthreads();
  }

  const int hi8 = (lane < 16) ? 0 : 8;
#pragma unroll
  for (int ni = 0; ni < 4; ++ni) {
    const int nc = n0 + wn * 64 + ni * 16 + (lane & 15);
    const float bn = bias[h * D_ + nc];
#pragma unroll
    for (int mi = 0; mi < 2; ++mi) {
#pragma unroll
      for (int r = 0; r < 8; ++r) {
        int m = m0 + wm * 32 + mi * 16 + r + hi8;    // global row = b*T + t
        int b = m >> 10, t = m & (T_ - 1);
        Y[(((size_t)b * H_ + h) * T_ + t) * D_ + nc] = f32_to_bf16(acc[mi][ni][r] + bn);
      }
    }
  }
}

// ---------------------------------------------------------------------------
// Flash attention: one block = 64 query rows of one (b,h). Online softmax.
// 8 waves. S-step: waves in 4x2 grid of 16x32 pieces. O accum: 64x768 f32 in
// registers, d-tile tt (48 tiles of 16) owned by wave (tt & 7).
// Q/K tiles staged with GLOBAL_LOAD_ASYNC_TO_LDS_B128 (pure bf16 copies).
// Writes Z bf16 in [B, T, H*D] layout for the fused output GEMM.
// ---------------------------------------------------------------------------
__global__ __launch_bounds__(256)
void attn_flash(const unsigned short* __restrict__ Q,
                const unsigned short* __restrict__ K,
                const unsigned short* __restrict__ V,
                unsigned short* __restrict__ Z) {
  __shared__ alignas(16) unsigned short Qs[64 * 32];    // [m][d]
  __shared__ alignas(16) unsigned short Ks[64 * 32];    // [s][d]
  __shared__ alignas(16) unsigned short Ps[64 * 64];    // [m][s] bf16 probs
  __shared__ alignas(16) unsigned short Vts[192 * 64];  // [d_local][s] transposed
  __shared__ float pmaxS[64 * 2], psumS[64 * 2];
  __shared__ float mrun[64], lrun[64], alphaS[64], mnewS[64];

  const int bh = blockIdx.y;
  const int b = bh / H_, h = bh % H_;
  const int m0  = blockIdx.x * 64;
  const int tid = threadIdx.x, w = tid >> 5, lane = tid & 31;
  const int wMs = w >> 1, wNs = w & 1;       // S-step wave grid 4x2
  const int hi8 = (lane < 16) ? 0 : 8;
  const size_t base = (size_t)bh * T_ * D_;
  const float scale = 0.036084391824351615f; // 1/sqrt(E)  (reference uses E!)

  const v8f vzero = {0.f, 0.f, 0.f, 0.f, 0.f, 0.f, 0.f, 0.f};
  v8f acc[6][4];                              // [owned d-tile][M tile]
#pragma unroll
  for (int q = 0; q < 6; ++q)
#pragma unroll
    for (int i = 0; i < 4; ++i) acc[q][i] = vzero;

  if (tid < 64) { mrun[tid] = -1e30f; lrun[tid] = 0.f; }
  __syncthreads();

  // per-thread 16B chunk for the 64x32 bf16 Q/K tiles (256 chunks each)
  const int ci  = tid >> 2;            // row 0..63
  const int cj8 = (tid & 3) * 8;       // col 0,8,16,24

  for (int s0 = 0; s0 < T_; s0 += 64) {
    // ---- S = scale * Q Kt  (64x64), accumulated over 24 d-chunks ----------
    v8f sacc[2] = {vzero, vzero};
    for (int d0 = 0; d0 < D_; d0 += 32) {
      async_copy_b128(&Q[base + (size_t)(m0 + ci) * D_ + d0 + cj8],
                      lds_addr32(&Qs[tid * 8]));
      async_copy_b128(&K[base + (size_t)(s0 + ci) * D_ + d0 + cj8],
                      lds_addr32(&Ks[tid * 8]));
      wait_async0();
      __syncthreads();
      v16bf a  = load_A(Qs, wMs * 16, 32);
      v16bf b0 = load_B(Ks, wNs * 32, 32);
      v16bf b1 = load_B(Ks, wNs * 32 + 16, 32);
      sacc[0] = wmma_bf16(a, b0, sacc[0]);
      sacc[1] = wmma_bf16(a, b1, sacc[1]);
      __syncthreads();
    }
    // ---- block row max (intra-wave shuffle, masks 1..8 stay in N-halves) --
    float rm[8];
#pragma unroll
    for (int r = 0; r < 8; ++r) {
      sacc[0][r] *= scale; sacc[1][r] *= scale;
      float v = fmaxf(sacc[0][r], sacc[1][r]);
      v = fmaxf(v, __shfl_xor(v, 1));
      v = fmaxf(v, __shfl_xor(v, 2));
      v = fmaxf(v, __shfl_xor(v, 4));
      v = fmaxf(v, __shfl_xor(v, 8));
      rm[r] = v;
    }
    if ((lane & 15) == 0) {
#pragma unroll
      for (int r = 0; r < 8; ++r) pmaxS[(wMs * 16 + r + hi8) * 2 + wNs] = rm[r];
    }
    __syncthreads();
    if (tid < 64) {
      float bm = fmaxf(pmaxS[tid * 2], pmaxS[tid * 2 + 1]);
      float mo = mrun[tid];
      float mn = fmaxf(mo, bm);
      mnewS[tid]  = mn;
      alphaS[tid] = __expf(mo - mn);
      mrun[tid]   = mn;
    }
    __syncthreads();
    // ---- P = exp(S - m_new) -> bf16 LDS, + partial row sums ---------------
#pragma unroll
    for (int r = 0; r < 8; ++r) {
      int m = wMs * 16 + r + hi8;
      float mn = mnewS[m];
      float p0 = __expf(sacc[0][r] - mn);
      float p1 = __expf(sacc[1][r] - mn);
      Ps[m * 64 + wNs * 32 + (lane & 15)]      = f32_to_bf16(p0);
      Ps[m * 64 + wNs * 32 + 16 + (lane & 15)] = f32_to_bf16(p1);
      float s = p0 + p1;
      s += __shfl_xor(s, 1); s += __shfl_xor(s, 2);
      s += __shfl_xor(s, 4); s += __shfl_xor(s, 8);
      if ((lane & 15) == 0) psumS[m * 2 + wNs] = s;
    }
    __syncthreads();
    if (tid < 64) lrun[tid] = lrun[tid] * alphaS[tid] + (psumS[tid * 2] + psumS[tid * 2 + 1]);
    // ---- rescale running O accumulators by alpha --------------------------
#pragma unroll
    for (int q = 0; q < 6; ++q)
#pragma unroll
      for (int i = 0; i < 4; ++i)
#pragma unroll
        for (int r = 0; r < 8; ++r)
          acc[q][i][r] *= alphaS[i * 16 + r + hi8];
    // ---- O += P * V, V streamed in 4 d-chunks of 192 cols -----------------
    for (int c = 0; c < 4; ++c) {
      __syncthreads();
      // 192x64 transposed fill, packed u32 stores along s (6144 pairs)
#pragma unroll
      for (int e = 0; e < 24; ++e) {
        int p = e * 256 + tid;               // (s2, dl): dl contiguous -> coalesced
        int s2 = p / 192, dl = p % 192;
        int s = s2 * 2;
        unsigned int lo = V[base + (size_t)(s0 + s) * D_ + c * 192 + dl];
        unsigned int hi = V[base + (size_t)(s0 + s + 1) * D_ + c * 192 + dl];
        *(unsigned int*)&Vts[dl * 64 + s] = lo | (hi << 16);
      }
      __syncthreads();
#pragma unroll
      for (int tloc = 0; tloc < 12; ++tloc) {
        int tt = c * 12 + tloc;
        if ((tt & 7) != w) continue;          // interleaved tile ownership
        int q = tt >> 3;
        v16bf b0 = load_B(Vts,      tloc * 16, 64);   // K = s 0..31
        v16bf b1 = load_B(Vts + 32, tloc * 16, 64);   // K = s 32..63
#pragma unroll
        for (int i = 0; i < 4; ++i) {
          v16bf a0 = load_A(Ps,      i * 16, 64);
          v16bf a1 = load_A(Ps + 32, i * 16, 64);
          acc[q][i] = wmma_bf16(a0, b0, acc[q][i]);
          acc[q][i] = wmma_bf16(a1, b1, acc[q][i]);
        }
      }
    }
    __syncthreads();
  }

  // ---- finalize: O / l  -> Z bf16 [b, t, h*D + d] --------------------------
  __syncthreads();
#pragma unroll
  for (int q = 0; q < 6; ++q) {
    int tt = w + 8 * q;
    int d = tt * 16 + (lane & 15);
#pragma unroll
    for (int i = 0; i < 4; ++i) {
#pragma unroll
      for (int r = 0; r < 8; ++r) {
        int m = i * 16 + r + hi8;
        int t = m0 + m;
        float val = acc[q][i][r] / lrun[m];
        Z[(((size_t)b * T_ + t) * (H_ * D_)) + (size_t)h * D_ + d] = f32_to_bf16(val);
      }
    }
  }
}

// ---------------------------------------------------------------------------
// Output projection: Out = Z(4096 x 6144, bf16) * Wo(6144 x 768) + sum_h bo
// grid (32, 6), block 256 (8 waves 4x2), tile 128x128, K step 32 (192 steps)
// A tile (already bf16) staged via async global->LDS b128.
// ---------------------------------------------------------------------------
__global__ __launch_bounds__(256)
void out_gemm(const unsigned short* __restrict__ Zb, const float* __restrict__ Wo,
              const float* __restrict__ bsum, float* __restrict__ Out) {
  __shared__ alignas(16) unsigned short As[128 * 32];
  __shared__ alignas(16) unsigned short Bs[128 * 32];
  const int m0  = blockIdx.x * 128;
  const int n0  = blockIdx.y * 128;
  const int tid = threadIdx.x;
  const int w = tid >> 5, lane = tid & 31;
  const int wm = w >> 1, wn = w & 1;
  const int KK = H_ * D_;   // 6144

  const v8f vzero = {0.f, 0.f, 0.f, 0.f, 0.f, 0.f, 0.f, 0.f};
  v8f acc[2][4];
#pragma unroll
  for (int mi = 0; mi < 2; ++mi)
#pragma unroll
    for (int ni = 0; ni < 4; ++ni) acc[mi][ni] = vzero;

  for (int k0 = 0; k0 < KK; k0 += 32) {
    if (k0 + 32 < KK) {
      __builtin_prefetch(&Zb[(size_t)(m0 + (tid >> 1)) * KK + k0 + 32], 0, 1);
      __builtin_prefetch(&Wo[(size_t)(k0 + 32 + (tid & 31)) * E_ + n0], 0, 1);
    }
    // A: 128x32 bf16 pure copy -> async to LDS (512 x 16B chunks)
#pragma unroll
    for (int e = 0; e < 2; ++e) {
      int cc = e * 256 + tid;
      int i = cc >> 2, j8 = (cc & 3) * 8;
      async_copy_b128(&Zb[(size_t)(m0 + i) * KK + k0 + j8],
                      lds_addr32(&As[cc * 8]));
    }
    // B: 32x128 f32 -> bf16 transposed [n][k], packed along k
#pragma unroll
    for (int e = 0; e < 8; ++e) {
      int p = e * 256 + tid;
      int n = p & 127, kk = (p >> 7) * 2;
      float blo = Wo[(size_t)(k0 + kk) * E_ + n0 + n];
      float bhi = Wo[(size_t)(k0 + kk + 1) * E_ + n0 + n];
      *(unsigned int*)&Bs[n * 32 + kk] = pack_bf16x2(blo, bhi);
    }
    wait_async0();
    __syncthreads();
    v16bf afr[2], bfr[4];
#pragma unroll
    for (int mi = 0; mi < 2; ++mi) afr[mi] = load_A(As, wm * 32 + mi * 16, 32);
#pragma unroll
    for (int ni = 0; ni < 4; ++ni) bfr[ni] = load_B(Bs, wn * 64 + ni * 16, 32);
#pragma unroll
    for (int mi = 0; mi < 2; ++mi)
#pragma unroll
      for (int ni = 0; ni < 4; ++ni)
        acc[mi][ni] = wmma_bf16(afr[mi], bfr[ni], acc[mi][ni]);
    __syncthreads();
  }

  const int hi8 = (lane < 16) ? 0 : 8;
#pragma unroll
  for (int ni = 0; ni < 4; ++ni) {
    const int nc = n0 + wn * 64 + ni * 16 + (lane & 15);
    const float bn = bsum[nc];
#pragma unroll
    for (int mi = 0; mi < 2; ++mi) {
#pragma unroll
      for (int r = 0; r < 8; ++r) {
        int m = m0 + wm * 32 + mi * 16 + r + hi8;
        Out[(size_t)m * E_ + nc] = acc[mi][ni][r] + bn;
      }
    }
  }
}

__global__ void bias_sum_k(const float* __restrict__ bo, float* __restrict__ bs) {
  int e = blockIdx.x * blockDim.x + threadIdx.x;
  if (e < E_) {
    float s = 0.f;
#pragma unroll
    for (int h = 0; h < H_; ++h) s += bo[h * E_ + e];
    bs[e] = s;
  }
}

// ---------------------------------------------------------------------------
extern "C" void kernel_launch(void* const* d_in, const int* in_sizes, int n_in,
                              void* d_out, int out_size, void* d_ws, size_t ws_size,
                              hipStream_t stream) {
  const float* x  = (const float*)d_in[0];
  const float* wq = (const float*)d_in[1];
  const float* bq = (const float*)d_in[2];
  const float* wk = (const float*)d_in[3];
  const float* bk = (const float*)d_in[4];
  const float* wv = (const float*)d_in[5];
  const float* bv = (const float*)d_in[6];
  const float* wo = (const float*)d_in[7];
  const float* bo = (const float*)d_in[8];

  const size_t N = (size_t)B_ * H_ * T_ * D_;      // 25,165,824 elements
  unsigned short* q = (unsigned short*)d_ws;
  unsigned short* k = q + N;
  unsigned short* v = k + N;
  unsigned short* z = v + N;
  float* bsum = (float*)(z + N);

  dim3 gproj(32, 6, H_);                           // 4096/128, 768/128, H
  proj_gemm<<<gproj, 256, 0, stream>>>(x, wq, bq, q);
  proj_gemm<<<gproj, 256, 0, stream>>>(x, wk, bk, k);
  proj_gemm<<<gproj, 256, 0, stream>>>(x, wv, bv, v);
  bias_sum_k<<<3, 256, 0, stream>>>(bo, bsum);

  attn_flash<<<dim3(T_ / 64, B_ * H_), 256, 0, stream>>>(q, k, v, z);

  out_gemm<<<dim3(32, 6), 256, 0, stream>>>(z, wo, bsum, (float*)d_out);
}
